// NavierStokesPINN1_63221918597227
// MI455X (gfx1250) — matrix operand order repaired and checked by reading, not compiled, not run on hardware
//
#include <hip/hip_runtime.h>

typedef _Float16 v8h  __attribute__((ext_vector_type(8)));
typedef _Float16 v16h __attribute__((ext_vector_type(16)));
typedef float    v8f  __attribute__((ext_vector_type(8)));

#define HDIM 256
#define LHID 4          // hidden->hidden layers (L-2)
#define ROWS 64         // points per workgroup
#define NRT  4          // row tiles (ROWS/16)
#define NTHREADS 512    // 16 wave32 waves; wave w owns col-tile w x all row-tiles

#define MIDX(L, RT, J, CT) ((((L) * NRT + (RT)) * 8 + (J)) * 16 + (CT))

// ---------------- prep: f32 weights -> f16 (row-major + transposed) -------
__global__ void prep_weights(const float* __restrict__ Wh,
                             _Float16* __restrict__ Whf,
                             _Float16* __restrict__ WhTf) {
    int idx = blockIdx.x * blockDim.x + threadIdx.x;
    if (idx >= LHID * HDIM * HDIM) return;
    int i = idx / (HDIM * HDIM);
    int k = (idx / HDIM) % HDIM;   // input-neuron (row of Wh[i])
    int n = idx % HDIM;            // output-neuron (col of Wh[i])
    float w = Wh[idx];
    Whf[idx] = (_Float16)w;                              // [i][k][n] row-major
    WhTf[i * HDIM * HDIM + n * HDIM + k] = (_Float16)w;  // [i][n][k] transposed
}

// ---------------- one 64x256 = 64x256 @ 256x256 GEMM stage ----------------
// src/dst: LDS activation buffers (ROWS x HDIM f16, row-major).
// Bmat: global f16, B-friendly layout: Bmat[n*HDIM + k] (contiguous along K).
// bias!=null => forward (ReLU + record ballot mask for layer `mlayer`);
// bias==null => backward (apply stored mask of layer `mlayer`).
__device__ __forceinline__ void gemm_layer(
    const _Float16* __restrict__ src, _Float16* __restrict__ dst,
    const _Float16* __restrict__ Bmat, const float* __restrict__ bias,
    unsigned* __restrict__ maskw, int mlayer, int lane, int wv) {
    const int ln15 = lane & 15;
    const int lhi  = (lane >> 4) & 1;
    const int ct   = wv;           // this wave's single column tile

    v8f acc[NRT] = {};             // 32 VGPRs of accumulators

    #pragma unroll
    for (int kc = 0; kc < 8; ++kc) {
        // B fragment 32x16 f16: lane holds col n, 16 contiguous K values.
        // Loaded once per kc, reused across all 4 row tiles.
        int n = ct * 16 + ln15;
        const _Float16* bp = Bmat + n * HDIM + kc * 32 + lhi * 16;
        v8h blo = *(const v8h*)bp;
        v8h bhi = *(const v8h*)(bp + 8);
        v16h b = __builtin_shufflevector(blo, bhi, 0, 1, 2, 3, 4, 5, 6, 7,
                                         8, 9, 10, 11, 12, 13, 14, 15);
        #pragma unroll
        for (int rt = 0; rt < NRT; ++rt) {
            // A fragment 16x32 f16: lane holds row m, 16 contiguous K values
            int abase = (rt * 16 + ln15) * HDIM + kc * 32 + lhi * 16;
            v8h alo = *(const v8h*)(src + abase);
            v8h ahi = *(const v8h*)(src + abase + 8);
            v16h a = __builtin_shufflevector(alo, ahi, 0, 1, 2, 3, 4, 5, 6, 7,
                                             8, 9, 10, 11, 12, 13, 14, 15);
            acc[rt] = __builtin_amdgcn_wmma_f32_16x16x32_f16(
                false, a, false, b, (short)0, acc[rt], false, false);
        }
    }

    // epilogue: bias+ReLU+ballot-mask (fwd) or mask-apply (bwd), store f16
    const int c = ct * 16 + ln15;
    float bc = bias ? bias[c] : 0.0f;
    #pragma unroll
    for (int rt = 0; rt < NRT; ++rt) {
        #pragma unroll
        for (int j = 0; j < 8; ++j) {
            int m = j + 8 * lhi;         // C/D row: VGPR j -> M=j / j+8
            float v = acc[rt][j] + bc;
            _Float16 hv;
            if (bias) {
                bool on = v > 0.0f;
                unsigned bal = __builtin_amdgcn_ballot_w32(on);
                if (lane == 0) maskw[MIDX(mlayer, rt, j, ct)] = bal;
                hv = on ? (_Float16)v : (_Float16)0.0f;
            } else {
                unsigned w = maskw[MIDX(mlayer, rt, j, ct)];
                bool on = (w >> lane) & 1u;
                hv = on ? (_Float16)v : (_Float16)0.0f;
            }
            dst[(rt * 16 + m) * HDIM + c] = hv;
        }
    }
}

// ---------------- fused PINN forward + 2-cotangent VJP --------------------
__launch_bounds__(NTHREADS)
__global__ void pinn_fused(const float* __restrict__ x, const float* __restrict__ y,
                           const float* __restrict__ t, const float* __restrict__ Win,
                           const float* __restrict__ b_in, const float* __restrict__ b_h,
                           const float* __restrict__ Wout, const float* __restrict__ b_out,
                           const _Float16* __restrict__ Whf,
                           const _Float16* __restrict__ WhTf,
                           float* __restrict__ out, int N) {
    __shared__ __attribute__((aligned(16))) _Float16 bufA[ROWS * HDIM]; // 32 KB
    __shared__ __attribute__((aligned(16))) _Float16 bufB[ROWS * HDIM]; // 32 KB
    __shared__ unsigned maskw[5 * NRT * 8 * 16];                        // 10 KB

    const int tid   = threadIdx.x;
    const int lane  = tid & 31;
    const int wv    = tid >> 5;        // wave 0..15 -> col tile
    const int ln15  = lane & 15;
    const int lhi   = (lane >> 4) & 1;
    const int gbase = blockIdx.x * ROWS;

    // ---- layer 1: h1 = relu(z @ Win + b_in), computed in C-fragment order
    {
        int c  = wv * 16 + ln15;
        float w0 = Win[c], w1 = Win[HDIM + c], w2 = Win[2 * HDIM + c];
        float bc = b_in[c];
        for (int rt = 0; rt < NRT; ++rt) {
            #pragma unroll
            for (int j = 0; j < 8; ++j) {
                int m  = j + 8 * lhi;
                int rg = gbase + rt * 16 + m;
                if (rg >= N) rg = N - 1;
                float a = x[rg] * w0 + y[rg] * w1 + t[rg] * w2 + bc;
                bool on = a > 0.0f;
                unsigned bal = __builtin_amdgcn_ballot_w32(on);
                if (lane == 0) maskw[MIDX(0, rt, j, wv)] = bal;
                bufA[(rt * 16 + m) * HDIM + c] = on ? (_Float16)a : (_Float16)0.0f;
            }
        }
    }
    __syncthreads();

    // ---- hidden layers 2..5 (WMMA GEMMs, ping-pong A<->B), h5 ends in bufA
    gemm_layer(bufA, bufB, WhTf + 0 * HDIM * HDIM, b_h + 0 * HDIM, maskw, 1, lane, wv);
    __syncthreads();
    gemm_layer(bufB, bufA, WhTf + 1 * HDIM * HDIM, b_h + 1 * HDIM, maskw, 2, lane, wv);
    __syncthreads();
    gemm_layer(bufA, bufB, WhTf + 2 * HDIM * HDIM, b_h + 2 * HDIM, maskw, 3, lane, wv);
    __syncthreads();
    gemm_layer(bufB, bufA, WhTf + 3 * HDIM * HDIM, b_h + 3 * HDIM, maskw, 4, lane, wv);
    __syncthreads();

    // ---- p = h5 @ Wout[:,1] + b_out[1]
    if (tid < ROWS) {
        float acc = 0.0f;
        for (int n = 0; n < HDIM; ++n)
            acc += (float)bufA[tid * HDIM + n] * Wout[n * 2 + 1];
        int idx = gbase + tid;
        if (idx < N) out[2 * N + idx] = acc + b_out[1];
    }

    // ---- backward VJP for cotangent cc: 0 = psi, 1 = p
    for (int cc = 0; cc < 2; ++cc) {
        // e5 = mask5 * Wout[:,cc]  -> bufB
        {
            int c  = wv * 16 + ln15;
            float wv5 = Wout[c * 2 + cc];
            for (int rt = 0; rt < NRT; ++rt) {
                #pragma unroll
                for (int j = 0; j < 8; ++j) {
                    int m = j + 8 * lhi;
                    unsigned w = maskw[MIDX(4, rt, j, wv)];
                    bool on = (w >> lane) & 1u;
                    bufB[(rt * 16 + m) * HDIM + c] = on ? (_Float16)wv5 : (_Float16)0.0f;
                }
            }
        }
        __syncthreads();  // also fences the p-reads of bufA above (cc==0)

        // e_{k-1} = mask_{k-1} * (e_k @ Wh[k-1]^T); B-friendly source = row-major Whf
        gemm_layer(bufB, bufA, Whf + 3 * HDIM * HDIM, nullptr, maskw, 3, lane, wv);
        __syncthreads();
        gemm_layer(bufA, bufB, Whf + 2 * HDIM * HDIM, nullptr, maskw, 2, lane, wv);
        __syncthreads();
        gemm_layer(bufB, bufA, Whf + 1 * HDIM * HDIM, nullptr, maskw, 1, lane, wv);
        __syncthreads();
        gemm_layer(bufA, bufB, Whf + 0 * HDIM * HDIM, nullptr, maskw, 0, lane, wv);
        __syncthreads();

        // gz[d] = e1 @ Win^T[:,d];  u=gz_psi[1], v=-gz_psi[0], f=gz_p[0], g=gz_p[1]
        if (tid < 2 * ROWS) {
            int r = tid >> 1, d = tid & 1;
            float acc = 0.0f;
            for (int n = 0; n < HDIM; ++n)
                acc += (float)bufB[r * HDIM + n] * Win[d * HDIM + n];
            int idx = gbase + r;
            if (idx < N) {
                if (cc == 0) {
                    if (d == 1) out[idx] = acc;          // u
                    else        out[N + idx] = -acc;     // v
                } else {
                    if (d == 0) out[3 * N + idx] = acc;  // f = p_x
                    else        out[4 * N + idx] = acc;  // g = p_y
                }
            }
        }
        __syncthreads();
    }
}

extern "C" void kernel_launch(void* const* d_in, const int* in_sizes, int n_in,
                              void* d_out, int out_size, void* d_ws, size_t ws_size,
                              hipStream_t stream) {
    const float* x     = (const float*)d_in[0];
    const float* y     = (const float*)d_in[1];
    const float* t     = (const float*)d_in[2];
    const float* Win   = (const float*)d_in[3];
    const float* b_in  = (const float*)d_in[4];
    const float* Wh    = (const float*)d_in[5];
    const float* b_h   = (const float*)d_in[6];
    const float* Wout  = (const float*)d_in[7];
    const float* b_out = (const float*)d_in[8];
    float* out = (float*)d_out;
    const int N = in_sizes[0];

    _Float16* Whf  = (_Float16*)d_ws;              // 4*256*256 f16 = 512 KB
    _Float16* WhTf = Whf + LHID * HDIM * HDIM;     // 512 KB more

    prep_weights<<<(LHID * HDIM * HDIM + 255) / 256, 256, 0, stream>>>(Wh, Whf, WhTf);

    int nblocks = (N + ROWS - 1) / ROWS;
    pinn_fused<<<nblocks, NTHREADS, 0, stream>>>(x, y, t, Win, b_in, b_h, Wout,
                                                 b_out, Whf, WhTf, out, N);
}
